// LatentModel_56822417326672
// MI455X (gfx1250) — matrix operand order repaired and checked by reading, not compiled, not run on hardware
//
#include <hip/hip_runtime.h>
#include <hip/hip_bf16.h>
#include <math.h>

typedef __bf16 bf16;
typedef __attribute__((ext_vector_type(16))) __bf16 v16bf;
typedef __attribute__((ext_vector_type(8)))  __bf16 bf16x8;
typedef __attribute__((ext_vector_type(8)))  float   v8f;

#define S_  256
#define B_  1024
#define D_  64
#define Z_  32
#define C_  64
#define H_  128
#define L_  515   // 2*S+3

static __device__ __forceinline__ v8f wmma_bf16(v16bf a, v16bf b, v8f c) {
  // D = A(16x32 bf16) * B(32x16 bf16) + C(16x16 f32)
  return __builtin_amdgcn_wmma_f32_16x16x32_bf16(false, a, false, b, (short)0, c,
                                                 false, false);
}

// ---- gfx1250 async global->LDS copy (ASYNCcnt-tracked) ---------------------
static __device__ __forceinline__ unsigned lds_off_of(const void* p) {
  // generic LDS pointers carry the LDS byte offset in the low 32 bits
  return (unsigned)(uintptr_t)p;
}
static __device__ __forceinline__ void async_copy_b128(unsigned lds_byte_off,
                                                       const void* gaddr) {
  asm volatile("global_load_async_to_lds_b128 %0, %1, off"
               :: "v"(lds_byte_off), "v"(gaddr) : "memory");
}
static __device__ __forceinline__ void wait_async0() {
  asm volatile("s_wait_asynccnt 0" ::: "memory");
}

// A fragment: 16x32 bf16 tile, row-major, row stride lda elements; p = &A[0][k0].
// Lane l: row m = l&15, half h = l>>4. elems 0..7 -> K = h*8+e ; 8..15 -> K = 16+h*8+(e-8).
static __device__ __forceinline__ v16bf load_a_frag(const bf16* p, int lda) {
  const int l  = threadIdx.x & 31;
  const int m  = l & 15;
  const int hh = l >> 4;
  const bf16* q = p + m * lda + hh * 8;
  union { v16bf v; bf16x8 h2[2]; } u;
  u.h2[0] = *(const bf16x8*)(q);
  u.h2[1] = *(const bf16x8*)(q + 16);
  return u.v;
}

// Same A-fragment, but source tile is f32 in LDS (async-copied raw); convert on the fly.
static __device__ __forceinline__ v16bf load_a_frag_f32cvt(const float* p, int lda) {
  const int l  = threadIdx.x & 31;
  const int m  = l & 15;
  const int hh = l >> 4;
  const float* q = p + m * lda + hh * 8;
  v16bf r;
#pragma unroll
  for (int e = 0; e < 8; ++e) r[e] = (bf16)q[e];
#pragma unroll
  for (int e = 0; e < 8; ++e) r[8 + e] = (bf16)q[16 + e];
  return r;
}

// B fragment for D = A * W^T where W is row-major [N][K] with stride ldw.
// B[k][n] = W[n][k]. Lane l: col n = l&15, half h = l>>4; elems 0..7 -> K = h*8+e ; 8..15 -> K = 16+h*8+(e-8).
static __device__ __forceinline__ v16bf load_b_frag(const bf16* p, int ldw) {
  const int l  = threadIdx.x & 31;
  const int n  = l & 15;
  const int hh = l >> 4;
  const bf16* q = p + n * ldw + hh * 8;
  union { v16bf v; bf16x8 h2[2]; } u;
  u.h2[0] = *(const bf16x8*)(q);
  u.h2[1] = *(const bf16x8*)(q + 16);
  return u.v;
}

// ---------------------------------------------------------------------------
// GRU backward-scan kernel. 64 blocks x 96 threads (3 waves).
// Block owns 16 batch rows for the full 256-step scan. Wave g in {0,1,2} owns
// gate chunk g (r/z/n). Weights LDS-resident (bf16); x tiles double-buffered
// via async global->LDS copies overlapping each step's WMMA work.
// ---------------------------------------------------------------------------
__global__ __launch_bounds__(96) void gru_scan_kernel(
    const float* __restrict__ xs,  const float* __restrict__ Wih,
    const float* __restrict__ Whh, const float* __restrict__ bih,
    const float* __restrict__ bhh, float* __restrict__ hT)
{
  __shared__ alignas(16) bf16  sWih[384 * 64];
  __shared__ alignas(16) bf16  sWhh[384 * 128];
  __shared__ alignas(16) float sXf [2][16 * 64];   // async-filled raw f32 tiles
  __shared__ alignas(16) bf16  sHb [16 * 128];
  __shared__ alignas(16) float sHf [16 * 128];
  __shared__ float sR [16 * 128];
  __shared__ float sZg[16 * 128];

  const int tid  = threadIdx.x;
  const int wave = tid >> 5;     // 0=r, 1=z, 2=n
  const int lane = tid & 31;
  const int n    = lane & 15;
  const int hh   = lane >> 4;
  const int b0   = blockIdx.x * 16;

  // kick off async copy of x_rev[0] tile while we stage weights
  {
    const float* x0 = xs + (size_t)(S_ - 1) * (B_ * D_) + (size_t)b0 * D_;
    unsigned base = lds_off_of(&sXf[0][0]);
    for (int byt = tid * 16; byt < 4096; byt += 96 * 16)
      async_copy_b128(base + byt, (const char*)x0 + byt);
  }

  for (int i = tid; i < 384 * 64;  i += 96) sWih[i] = (bf16)Wih[i];
  for (int i = tid; i < 384 * 128; i += 96) sWhh[i] = (bf16)Whh[i];
  for (int i = tid; i < 16 * 128;  i += 96) { sHf[i] = 0.f; sHb[i] = (bf16)0.f; }

  float bi[8], bh[8];
#pragma unroll
  for (int t = 0; t < 8; ++t) {
    bi[t] = bih[wave * 128 + t * 16 + n];
    bh[t] = bhh[wave * 128 + t * 16 + n];
  }

  for (int s = 0; s < S_; ++s) {
    wait_async0();       // this wave's portion of tile s landed
    __syncthreads();     // every wave's portion landed (also covers weight staging)

    // prefetch next x tile into the other buffer; overlaps this whole step
    if (s + 1 < S_) {
      const float* xn = xs + (size_t)(S_ - 2 - s) * (B_ * D_) + (size_t)b0 * D_;
      unsigned base = lds_off_of(&sXf[(s + 1) & 1][0]);
      for (int byt = tid * 16; byt < 4096; byt += 96 * 16)
        async_copy_b128(base + byt, (const char*)xn + byt);
    }
    const float* xcur = &sXf[s & 1][0];

    v8f ai[8], ah[8];
#pragma unroll
    for (int t = 0; t < 8; ++t) {
      v8f c = {};
#pragma unroll
      for (int v = 0; v < 8; ++v) c[v] = bi[t];
#pragma unroll
      for (int kk = 0; kk < 2; ++kk) {   // gi: K=64
        v16bf a = load_a_frag_f32cvt(xcur + kk * 32, 64);
        v16bf b = load_b_frag(sWih + (wave * 128 + t * 16) * 64 + kk * 32, 64);
        c = wmma_bf16(a, b, c);
      }
      ai[t] = c;
      v8f ch = {};
#pragma unroll
      for (int v = 0; v < 8; ++v) ch[v] = bh[t];
#pragma unroll
      for (int kk = 0; kk < 4; ++kk) {   // gh: K=128
        v16bf a = load_a_frag(sHb + kk * 32, 128);
        v16bf b = load_b_frag(sWhh + (wave * 128 + t * 16) * 128 + kk * 32, 128);
        ch = wmma_bf16(a, b, ch);
      }
      ah[t] = ch;
    }

    if (wave == 0) {           // r = sigmoid(i_r + h_r)
#pragma unroll
      for (int t = 0; t < 8; ++t)
#pragma unroll
        for (int v = 0; v < 8; ++v) {
          int m = v + 8 * hh;
          float x = ai[t][v] + ah[t][v];
          sR[m * 128 + t * 16 + n] = 1.f / (1.f + expf(-x));
        }
    } else if (wave == 1) {    // zg = sigmoid(i_z + h_z)
#pragma unroll
      for (int t = 0; t < 8; ++t)
#pragma unroll
        for (int v = 0; v < 8; ++v) {
          int m = v + 8 * hh;
          float x = ai[t][v] + ah[t][v];
          sZg[m * 128 + t * 16 + n] = 1.f / (1.f + expf(-x));
        }
    }
    __syncthreads();
    if (wave == 2) {           // n = tanh(i_n + r*h_n); h = (1-zg)*n + zg*h
#pragma unroll
      for (int t = 0; t < 8; ++t)
#pragma unroll
        for (int v = 0; v < 8; ++v) {
          int m   = v + 8 * hh;
          int col = t * 16 + n;
          float r   = sR [m * 128 + col];
          float zg  = sZg[m * 128 + col];
          float nn  = tanhf(ai[t][v] + r * ah[t][v]);
          float hnw = (1.f - zg) * nn + zg * sHf[m * 128 + col];
          sHf[m * 128 + col] = hnw;
          sHb[m * 128 + col] = (bf16)hnw;
        }
    }
    __syncthreads();
  }
  for (int i = tid; i < 16 * 128; i += 96) hT[(size_t)b0 * 128 + i] = sHf[i];
}

// ---------------------------------------------------------------------------
// Encoder: ctx0 = hT@encW^T+b ; q = ctx0@qW^T+b ; z0 = mean + exp(logstd)*eps
// plus per-batch-row KL partial. 1024 blocks x 64 threads (tiny phase).
// ---------------------------------------------------------------------------
__global__ __launch_bounds__(64) void enc_kernel(
    const float* __restrict__ hT,  const float* __restrict__ encW,
    const float* __restrict__ encb, const float* __restrict__ qW,
    const float* __restrict__ qb,  const float* __restrict__ eps,
    const float* __restrict__ pzm, const float* __restrict__ pzl,
    float* __restrict__ z0, float* __restrict__ klp)
{
  __shared__ float sh[128];
  __shared__ float sc[64];
  __shared__ float sq[64];
  __shared__ float skl[32];
  const int b = blockIdx.x, t = threadIdx.x;
  sh[t]      = hT[(size_t)b * 128 + t];
  sh[t + 64] = hT[(size_t)b * 128 + 64 + t];
  __syncthreads();
  float acc = encb[t];
  for (int k = 0; k < 128; ++k) acc += encW[t * 128 + k] * sh[k];
  sc[t] = acc;
  __syncthreads();
  float q = qb[t];
  for (int k = 0; k < 64; ++k) q += qW[t * 64 + k] * sc[k];
  sq[t] = q;
  __syncthreads();
  if (t < 32) {
    float mean = sq[t], lst = sq[32 + t];
    z0[(size_t)b * 32 + t] = mean + expf(lst) * eps[(size_t)b * 32 + t];
    float pl = pzl[t], pm = pzm[t];
    float dm = mean - pm;
    skl[t] = pl - lst + (expf(2.f * lst) + dm * dm) / (2.f * expf(2.f * pl)) - 0.5f;
  }
  __syncthreads();
  if (t == 0) {
    float s = 0.f;
    for (int i = 0; i < 32; ++i) s += skl[i];
    klp[b] = s;
  }
}

// ---------------------------------------------------------------------------
// SDE scan, 514 steps, fused with projection / MSE / interp output.
// 64 blocks x 128 threads (4 waves). Block owns 16 batch rows.
// dW tiles double-buffered via async global->LDS copies one step ahead.
// ---------------------------------------------------------------------------
__global__ __launch_bounds__(128) void sde_scan_kernel(
    const float* __restrict__ ext_ts, const float* __restrict__ dWn,
    const float* __restrict__ z0,
    const float* __restrict__ fW1, const float* __restrict__ fb1,
    const float* __restrict__ fW2, const float* __restrict__ fb2,
    const float* __restrict__ fW3, const float* __restrict__ fb3,
    const float* __restrict__ gW1, const float* __restrict__ gb1,
    const float* __restrict__ gW2, const float* __restrict__ gb2,
    const float* __restrict__ pW,  const float* __restrict__ pb,
    const float* __restrict__ xs,  float* __restrict__ interp,
    float* __restrict__ sq_partials)
{
  __shared__ alignas(16) bf16 sFW1[128 * 64];   // K padded 33->64 with zeros
  __shared__ alignas(16) bf16 sGW1[128 * 64];
  __shared__ alignas(16) bf16 sFW2[128 * 128];
  __shared__ alignas(16) bf16 sFW3[32 * 128];
  __shared__ alignas(16) bf16 sGW2[32 * 128];
  __shared__ alignas(16) bf16 sPW [64 * 64];    // col0 zero (t slot), cols 1..32 = proj_W
  __shared__ alignas(16) bf16 sTZ [16 * 64];    // [t, z(32), zero pad]
  __shared__ alignas(16) bf16 sH1 [16 * 128];
  __shared__ alignas(16) bf16 sGH [16 * 128];
  __shared__ alignas(16) bf16 sH2 [16 * 128];
  __shared__ alignas(16) float sDW[2][16 * 32]; // async-filled dW tiles
  __shared__ float sDiff[16 * 32];
  __shared__ float sZ  [16 * 32];
  __shared__ float sRed[128];

  const int tid  = threadIdx.x;
  const int wave = tid >> 5;
  const int lane = tid & 31;
  const int n    = lane & 15;
  const int hh   = lane >> 4;
  const int b0   = blockIdx.x * 16;

  // kick off async copy of dW[0] tile (2KB: one b128 per thread)
  {
    unsigned base = lds_off_of(&sDW[0][0]);
    async_copy_b128(base + tid * 16,
                    (const char*)(dWn + (size_t)b0 * Z_) + tid * 16);
  }

  for (int i = tid; i < 128 * 64; i += 128) {
    int r = i >> 6, k = i & 63;
    sFW1[i] = (k < 33) ? (bf16)fW1[r * 33 + k] : (bf16)0.f;
    sGW1[i] = (k < 33) ? (bf16)gW1[r * 33 + k] : (bf16)0.f;
  }
  for (int i = tid; i < 128 * 128; i += 128) sFW2[i] = (bf16)fW2[i];
  for (int i = tid; i < 32 * 128;  i += 128) { sFW3[i] = (bf16)fW3[i]; sGW2[i] = (bf16)gW2[i]; }
  for (int i = tid; i < 64 * 64;   i += 128) {
    int r = i >> 6, k = i & 63;
    sPW[i] = (k >= 1 && k < 33) ? (bf16)pW[r * 32 + (k - 1)] : (bf16)0.f;
  }
  {
    float t0v = ext_ts[0];
    for (int i = tid; i < 16 * 32; i += 128) sZ[i] = z0[(size_t)b0 * 32 + i];
    for (int i = tid; i < 16 * 64; i += 128) {
      int r = i >> 6, k = i & 63;
      float v = (k == 0) ? t0v : ((k < 33) ? z0[(size_t)b0 * 32 + r * 32 + (k - 1)] : 0.f);
      sTZ[i] = (bf16)v;
    }
  }

  float bL1[4];
#pragma unroll
  for (int j = 0; j < 4; ++j)
    bL1[j] = (wave < 2) ? fb1[(wave * 4 + j) * 16 + n] : gb1[((wave - 2) * 4 + j) * 16 + n];
  float bL2[4];
#pragma unroll
  for (int j = 0; j < 4; ++j)
    bL2[j] = (wave < 2) ? fb2[(wave * 4 + j) * 16 + n]
                        : ((wave == 2 && j < 2) ? gb2[j * 16 + n] : 0.f);
  const float bL3 = (wave < 2) ? fb3[wave * 16 + n] : 0.f;
  const float bP  = pb[wave * 16 + n];

  float acc_sq = 0.f;
  __syncthreads();

  for (int st = 0; st < L_ - 1; ++st) {
    const float t1   = ext_ts[st + 1];
    const float dt   = t1 - ext_ts[st];
    const float sqdt = sqrtf(dt);

    // ---- Layer 1: h1 = relu(tz@fW1^T) (waves 0,1) ; gh = relu(tz@gW1^T) (waves 2,3)
    {
      const bf16* W  = (wave < 2) ? sFW1 : sGW1;
      bf16*      dst = (wave < 2) ? sH1 : sGH;
      const int base_t = (wave & 1) * 4;
#pragma unroll
      for (int j = 0; j < 4; ++j) {
        const int tt = base_t + j;
        v8f c = {};
#pragma unroll
        for (int v = 0; v < 8; ++v) c[v] = bL1[j];
#pragma unroll
        for (int kk = 0; kk < 2; ++kk) {
          v16bf a = load_a_frag(sTZ + kk * 32, 64);
          v16bf b = load_b_frag(W + (tt * 16) * 64 + kk * 32, 64);
          c = wmma_bf16(a, b, c);
        }
#pragma unroll
        for (int v = 0; v < 8; ++v) {
          int m = v + 8 * hh;
          float x = c[v] > 0.f ? c[v] : 0.f;
          dst[m * 128 + tt * 16 + n] = (bf16)x;
        }
      }
    }
    __syncthreads();
    // ---- Layer 2: h2 = relu(h1@fW2^T) (waves 0,1) ; diff = gh@gW2^T + b (wave 2)
    if (wave < 2) {
#pragma unroll
      for (int j = 0; j < 4; ++j) {
        const int tt = wave * 4 + j;
        v8f c = {};
#pragma unroll
        for (int v = 0; v < 8; ++v) c[v] = bL2[j];
#pragma unroll
        for (int kk = 0; kk < 4; ++kk) {
          v16bf a = load_a_frag(sH1 + kk * 32, 128);
          v16bf b = load_b_frag(sFW2 + (tt * 16) * 128 + kk * 32, 128);
          c = wmma_bf16(a, b, c);
        }
#pragma unroll
        for (int v = 0; v < 8; ++v) {
          int m = v + 8 * hh;
          float x = c[v] > 0.f ? c[v] : 0.f;
          sH2[m * 128 + tt * 16 + n] = (bf16)x;
        }
      }
    } else if (wave == 2) {
#pragma unroll
      for (int j = 0; j < 2; ++j) {
        v8f c = {};
#pragma unroll
        for (int v = 0; v < 8; ++v) c[v] = bL2[j];
#pragma unroll
        for (int kk = 0; kk < 4; ++kk) {
          v16bf a = load_a_frag(sGH + kk * 32, 128);
          v16bf b = load_b_frag(sGW2 + (j * 16) * 128 + kk * 32, 128);
          c = wmma_bf16(a, b, c);
        }
#pragma unroll
        for (int v = 0; v < 8; ++v) {
          int m = v + 8 * hh;
          sDiff[m * 32 + j * 16 + n] = c[v];
        }
      }
    }
    wait_async0();       // dW tile for this step landed (issued one step ago)
    __syncthreads();
    // ---- Layer 3 + Euler-Maruyama update (waves 0,1); wave 3 refreshes t column
    if (wave < 2) {
      v8f c = {};
#pragma unroll
      for (int v = 0; v < 8; ++v) c[v] = bL3;
#pragma unroll
      for (int kk = 0; kk < 4; ++kk) {
        v16bf a = load_a_frag(sH2 + kk * 32, 128);
        v16bf b = load_b_frag(sFW3 + (wave * 16) * 128 + kk * 32, 128);
        c = wmma_bf16(a, b, c);
      }
      const float* dwt = &sDW[st & 1][0];
#pragma unroll
      for (int v = 0; v < 8; ++v) {
        int m  = v + 8 * hh;
        int zc = wave * 16 + n;
        float zold = sZ[m * 32 + zc];
        float dwv  = dwt[m * 32 + zc];
        float znew = zold + c[v] * dt + sDiff[m * 32 + zc] * (sqdt * dwv);
        sZ[m * 32 + zc]       = znew;
        sTZ[m * 64 + 1 + zc]  = (bf16)znew;
      }
    } else if (wave == 3 && lane < 16) {
      sTZ[lane * 64] = (bf16)t1;   // next step's time column (proj weight col0 is zero)
    }
    // prefetch next dW tile into the other buffer (overlaps proj + next L1/L2)
    if (st + 1 < L_ - 1) {
      unsigned base = lds_off_of(&sDW[(st + 1) & 1][0]);
      async_copy_b128(base + tid * 16,
                      (const char*)(dWn + (size_t)(st + 1) * (B_ * Z_) +
                                    (size_t)b0 * Z_) + tid * 16);
    }
    __syncthreads();
    // ---- Fused projection xhat_l = z_l @ proj^T + b for l = st+1
    const int l = st + 1;
    const bool isRecon  = ((l & 1) == 0) && (l >= 2) && (l <= 512);
    const bool isInterp = ((l & 1) == 1) && (l >= 3) && (l <= 511);
    if (isRecon || isInterp) {
      v8f c = {};
#pragma unroll
      for (int v = 0; v < 8; ++v) c[v] = bP;
#pragma unroll
      for (int kk = 0; kk < 2; ++kk) {
        v16bf a = load_a_frag(sTZ + kk * 32, 64);
        v16bf b = load_b_frag(sPW + (wave * 16) * 64 + kk * 32, 64);
        c = wmma_bf16(a, b, c);
      }
#pragma unroll
      for (int v = 0; v < 8; ++v) {
        int m = v + 8 * hh;
        int d = wave * 16 + n;
        if (isRecon) {
          float ref = xs[(size_t)((l - 2) >> 1) * (B_ * D_) + (size_t)(b0 + m) * D_ + d];
          float df  = c[v] - ref;
          acc_sq += df * df;
        } else {
          interp[(size_t)((l - 3) >> 1) * (B_ * D_) + (size_t)(b0 + m) * D_ + d] = c[v];
        }
      }
      if (isRecon && l + 2 <= 512) {  // warm cache for next recon compare tile
        const char* nx = (const char*)(xs + (size_t)(l >> 1) * (B_ * D_) + (size_t)b0 * D_);
        __builtin_prefetch(nx + tid * 32, 0, 0);
      }
    }
    __syncthreads();
  }

  sRed[tid] = acc_sq;
  __syncthreads();
  for (int off = 64; off > 0; off >>= 1) {
    if (tid < off) sRed[tid] += sRed[tid + off];
    __syncthreads();
  }
  if (tid == 0) sq_partials[blockIdx.x] = sRed[0];
}

// ---------------------------------------------------------------------------
__global__ void finalize_kernel(const float* __restrict__ sqp,
                                const float* __restrict__ klp,
                                float* __restrict__ out)
{
  if (threadIdx.x == 0 && blockIdx.x == 0) {
    double s = 0.0;
    for (int i = 0; i < 64; ++i) s += (double)sqp[i];
    out[0] = (float)(s / ((double)S_ * (double)B_ * (double)D_));
    double k = 0.0;
    for (int i = 0; i < B_; ++i) k += (double)klp[i];
    out[1] = (float)(k / (double)B_);
  }
}

// ---------------------------------------------------------------------------
extern "C" void kernel_launch(void* const* d_in, const int* in_sizes, int n_in,
                              void* d_out, int out_size, void* d_ws, size_t ws_size,
                              hipStream_t stream) {
  (void)in_sizes; (void)n_in; (void)out_size; (void)ws_size;
  const float* xs      = (const float*)d_in[0];
  const float* ext_ts  = (const float*)d_in[1];
  const float* eps_z0  = (const float*)d_in[2];
  const float* dWn     = (const float*)d_in[3];
  const float* gru_Wih = (const float*)d_in[4];
  const float* gru_Whh = (const float*)d_in[5];
  const float* gru_bih = (const float*)d_in[6];
  const float* gru_bhh = (const float*)d_in[7];
  const float* enc_W   = (const float*)d_in[8];
  const float* enc_b   = (const float*)d_in[9];
  const float* qz0_W   = (const float*)d_in[10];
  const float* qz0_b   = (const float*)d_in[11];
  const float* f_W1    = (const float*)d_in[12];
  const float* f_b1    = (const float*)d_in[13];
  const float* f_W2    = (const float*)d_in[14];
  const float* f_b2    = (const float*)d_in[15];
  const float* f_W3    = (const float*)d_in[16];
  const float* f_b3    = (const float*)d_in[17];
  const float* g_W1    = (const float*)d_in[18];
  const float* g_b1    = (const float*)d_in[19];
  const float* g_W2    = (const float*)d_in[20];
  const float* g_b2    = (const float*)d_in[21];
  const float* proj_W  = (const float*)d_in[22];
  const float* proj_b  = (const float*)d_in[23];
  const float* pz0_m   = (const float*)d_in[24];
  const float* pz0_l   = (const float*)d_in[25];

  float* ws  = (float*)d_ws;
  float* hT  = ws;                    // 1024*128
  float* z0  = ws + 131072;           // 1024*32
  float* sqp = ws + 131072 + 32768;   // 64
  float* klp = sqp + 64;              // 1024
  float* out = (float*)d_out;

  gru_scan_kernel<<<dim3(B_ / 16), dim3(96), 0, stream>>>(
      xs, gru_Wih, gru_Whh, gru_bih, gru_bhh, hT);
  enc_kernel<<<dim3(B_), dim3(64), 0, stream>>>(
      hT, enc_W, enc_b, qz0_W, qz0_b, eps_z0, pz0_m, pz0_l, z0, klp);
  sde_scan_kernel<<<dim3(B_ / 16), dim3(128), 0, stream>>>(
      ext_ts, dWn, z0, f_W1, f_b1, f_W2, f_b2, f_W3, f_b3,
      g_W1, g_b1, g_W2, g_b2, proj_W, proj_b, xs, out + 2, sqp);
  finalize_kernel<<<dim3(1), dim3(1), 0, stream>>>(sqp, klp, out);
}
